// LWRLayer_19456201851218
// MI455X (gfx1250) — compile-verified, matrix operand order checked by reading, not run on hardware
//
#include <hip/hip_runtime.h>

// ---------- CDNA5 WMMA types ----------
typedef __attribute__((ext_vector_type(16))) __bf16 v16bf;
typedef __attribute__((ext_vector_type(8)))  float  v8f;

#define INFF 3.4028234663852886e38f

#define QT   32      // queries per block (two WMMA M tiles)
#define NC   8192    // db rows per chunk (per block in grid.x)
#define PH   512     // db rows per phase (LDS staging granularity)
#define TPB  256     // 8 waves of 32
#define CAND 128     // candidates kept per (query, chunk) = 32 lanes * 4

// ---------------------------------------------------------------
// Kernel: convert X to bf16 (row-major) and compute ||x||^2
// ---------------------------------------------------------------
__global__ void x_prep(const float* __restrict__ X,
                       unsigned short* __restrict__ Xb,
                       float* __restrict__ xsq, int B) {
    int r = blockIdx.x * blockDim.x + threadIdx.x;
    if (r < B) {
        const float* xp = X + (size_t)r * 64;
        float s = 0.f;
        #pragma unroll
        for (int j = 0; j < 64; ++j) {
            float v = xp[j];
            s += v * v;
            union { __bf16 h; unsigned short u; } cv;
            cv.h = (__bf16)v;
            Xb[(size_t)r * 64 + j] = cv.u;
        }
        xsq[r] = s;
    }
}

// ---------------------------------------------------------------
// Kernel: one-pass db f32 -> bf16 conversion + per-row ||d||^2.
// 4 threads per row (16 floats each), shfl-reduce the square sum.
// ---------------------------------------------------------------
__global__ void db_prep(const float* __restrict__ db,
                        unsigned short* __restrict__ dbb,
                        float* __restrict__ dbsq, int Nn) {
    int t = blockIdx.x * blockDim.x + threadIdx.x;
    int row  = t >> 2;
    int part = t & 3;
    if (row < Nn) {
        const float* rp = db + (size_t)row * 64 + part * 16;
        unsigned short* op = dbb + (size_t)row * 64 + part * 16;
        float s = 0.f;
        #pragma unroll
        for (int j = 0; j < 16; ++j) {
            float v = rp[j];
            s += v * v;
            union { __bf16 h; unsigned short u; } cv;
            cv.h = (__bf16)v;
            op[j] = cv.u;
        }
        s += __shfl_xor(s, 1);
        s += __shfl_xor(s, 2);
        if (part == 0) dbsq[row] = s;
    }
}

// per-lane sorted top-4 insertion (ascending)
__device__ __forceinline__ void insert4(float d, int idx, float* td, int* ti) {
    if (d >= td[3]) return;
    if (d < td[1]) {
        td[3] = td[2]; ti[3] = ti[2];
        td[2] = td[1]; ti[2] = ti[1];
        if (d < td[0]) { td[1] = td[0]; ti[1] = ti[0]; td[0] = d; ti[0] = idx; }
        else           { td[1] = d;     ti[1] = idx; }
    } else {
        if (d < td[2]) { td[3] = td[2]; ti[3] = ti[2]; td[2] = d; ti[2] = idx; }
        else           { td[3] = d;     ti[3] = idx; }
    }
}

// ---------------------------------------------------------------
// Main kernel: fused WMMA distance GEMM + per-chunk candidate select
//   grid.x = chunk over N (NC rows each), grid.y = query tile (32 q)
//
// Selection operates on the monotone score s = ||d||^2 - 2 x.d ;
// the per-query affine map mse = (s + ||x||^2)/64 is applied only to
// the 128 surviving candidates at emit time (saves ~2/3 of the
// per-tile epilogue VALU work).
//
//   PRE = true : db pre-converted to bf16 (dbb) + dbsq precomputed
//   PRE = false: convert db f32 -> bf16 in-register (low-ws fallback)
// ---------------------------------------------------------------
template <bool PRE>
__global__ void __launch_bounds__(TPB)
knn_main(const unsigned short* __restrict__ Xb,
         const float* __restrict__ xsq,
         const float* __restrict__ dbf,            // f32 db (fallback path)
         const unsigned short* __restrict__ dbbu,  // bf16 db (PRE path)
         const float* __restrict__ dbsq,           // ||d||^2 (PRE path)
         float* __restrict__ cand_d,
         int*   __restrict__ cand_i,
         int Nn, int S) {
    __shared__ float s_dist[QT * PH];          // 64 KB score staging

    const int chunk = blockIdx.x;
    const int qt    = blockIdx.y;
    const int tid   = threadIdx.x;
    const int lane  = tid & 31;
    const int w     = tid >> 5;                // wave id, 0..7
    const int rsel  = lane & 15;               // row/col within 16
    const int khalf = (lane >> 4) & 1;         // K-half this lane packs
    const int mhalf = khalf * 8;               // C-layout: lanes 16..31 -> M=r+8

    const __bf16* xb  = reinterpret_cast<const __bf16*>(Xb);
    const __bf16* dbb = reinterpret_cast<const __bf16*>(dbbu);

    // --- A fragments: 2 M-tiles x K=64 ---
    v16bf a[2][2];
    #pragma unroll
    for (int mt = 0; mt < 2; ++mt) {
        const __bf16* xr = xb + ((size_t)(qt * QT + mt * 16 + rsel)) * 64;
        a[mt][0] = *(const v16bf*)(xr +  0 + khalf * 16);
        a[mt][1] = *(const v16bf*)(xr + 32 + khalf * 16);
    }

    // --- running per-lane top-4 (on score s) for the wave's 4 queries ---
    float td[4][4];
    int   ti[4][4];
    #pragma unroll
    for (int qq = 0; qq < 4; ++qq)
        #pragma unroll
        for (int j = 0; j < 4; ++j) { td[qq][j] = INFF; ti[qq][j] = -1; }

    const int phases = NC / PH;                // 16
    for (int ph = 0; ph < phases; ++ph) {
        const int phase_base = chunk * NC + ph * PH;

        // ------- GEMM phase: 32 N-tiles of 16 rows, 4 per wave -------
        #pragma unroll
        for (int t = 0; t < 4; ++t) {
            const int tilecol = w * 4 + t;               // 0..31
            const int n0 = phase_base + tilecol * 16;
            int row = n0 + rsel;                         // 32-bit clamp
            if (row >= Nn) row = Nn - 1;

            v16bf b0, b1;
            float ssq;
            if constexpr (PRE) {
                const __bf16* bp = dbb + (size_t)(unsigned)row * 64;
                __builtin_prefetch(bp + (size_t)PH * 64, 0, 1);
                b0 = *(const v16bf*)(bp + khalf * 16);
                b1 = *(const v16bf*)(bp + 32 + khalf * 16);
                ssq = dbsq[row];
            } else {
                const float* rp = dbf + (size_t)(unsigned)row * 64;
                __builtin_prefetch(rp + (size_t)PH * 64, 0, 1);
                float s = 0.f;
                #pragma unroll
                for (int j = 0; j < 16; ++j) {
                    float v = rp[khalf * 16 + j];
                    s += v * v;
                    b0[j] = (__bf16)v;
                }
                #pragma unroll
                for (int j = 0; j < 16; ++j) {
                    float v = rp[32 + khalf * 16 + j];
                    s += v * v;
                    b1[j] = (__bf16)v;
                }
                s += __shfl_xor(s, 16, 32);
                ssq = s;
            }

            // interleave the two accumulators so back-to-back WMMAs
            // are independent (hides WMMA->WMMA RAW hazard NOPs)
            v8f c0 = {}, c1 = {};
            c0 = __builtin_amdgcn_wmma_f32_16x16x32_bf16(false, a[0][0], false, b0,
                                                         (short)0, c0, false, false);
            c1 = __builtin_amdgcn_wmma_f32_16x16x32_bf16(false, a[1][0], false, b0,
                                                         (short)0, c1, false, false);
            c0 = __builtin_amdgcn_wmma_f32_16x16x32_bf16(false, a[0][1], false, b1,
                                                         (short)0, c0, false, false);
            c1 = __builtin_amdgcn_wmma_f32_16x16x32_bf16(false, a[1][1], false, b1,
                                                         (short)0, c1, false, false);

            // score s = ||d||^2 - 2 x.d  (one FMA per element), stage in LDS
            #pragma unroll
            for (int r = 0; r < 8; ++r) {
                float s0 = __builtin_fmaf(-2.0f, c0[r], ssq);
                float s1 = __builtin_fmaf(-2.0f, c1[r], ssq);
                s_dist[(     r + mhalf) * PH + tilecol * 16 + rsel] = s0;
                s_dist[(16 + r + mhalf) * PH + tilecol * 16 + rsel] = s1;
            }
        }
        __syncthreads();

        // ------- selection phase: wave w owns queries 4w .. 4w+3 -------
        #pragma unroll
        for (int qq = 0; qq < 4; ++qq) {
            const int q = w * 4 + qq;
            const float* sd = s_dist + q * PH;
            for (int i = lane; i < PH; i += 32) {
                int gidx = phase_base + i;
                float d = (gidx < Nn) ? sd[i] : INFF;
                insert4(d, gidx, td[qq], ti[qq]);
            }
        }
        __syncthreads();
    }

    // ------- emit 128 candidates per (query, chunk) -------
    // apply the per-query affine map mse = (s + ||x||^2) / 64 here
    #pragma unroll
    for (int qq = 0; qq < 4; ++qq) {
        const int qg = qt * QT + w * 4 + qq;
        const float xq = xsq[qg];
        size_t base = ((size_t)qg * S + chunk) * CAND + (size_t)lane * 4;
        #pragma unroll
        for (int j = 0; j < 4; ++j) {
            cand_d[base + j] = (td[qq][j] + xq) * (1.0f / 64.0f);
            cand_i[base + j] = ti[qq][j];
        }
    }
}

// ---------------------------------------------------------------
// Final: exact top-k merge over candidates (one block per query)
// ---------------------------------------------------------------
__global__ void __launch_bounds__(TPB)
topk_kernel(const float* __restrict__ cand_d,
            const int*   __restrict__ cand_i,
            float* __restrict__ out_d,
            int*   __restrict__ out_i,
            int C, int k) {
    extern __shared__ float sd[];              // C floats
    __shared__ float rv[TPB];
    __shared__ int   ri[TPB];

    const int q   = blockIdx.x;
    const int tid = threadIdx.x;
    const float* cd = cand_d + (size_t)q * C;

    for (int i = tid; i < C; i += TPB) sd[i] = cd[i];
    __syncthreads();

    for (int it = 0; it < k; ++it) {
        float mv = INFF; int mp = -1;
        for (int i = tid; i < C; i += TPB) {
            float v = sd[i];
            if (v < mv) { mv = v; mp = i; }
        }
        rv[tid] = mv; ri[tid] = mp;
        __syncthreads();
        #pragma unroll
        for (int off = TPB / 2; off > 0; off >>= 1) {
            if (tid < off && rv[tid + off] < rv[tid]) {
                rv[tid] = rv[tid + off];
                ri[tid] = ri[tid + off];
            }
            __syncthreads();
        }
        if (tid == 0) {
            int p = ri[0];
            out_d[(size_t)q * k + it] = rv[0];
            out_i[(size_t)q * k + it] = (p >= 0) ? cand_i[(size_t)q * C + p] : -1;
            if (p >= 0) sd[p] = INFF;
        }
        __syncthreads();
    }
}

// ---------------------------------------------------------------
extern "C" void kernel_launch(void* const* d_in, const int* in_sizes, int n_in,
                              void* d_out, int out_size, void* d_ws, size_t ws_size,
                              hipStream_t stream) {
    const float* X  = (const float*)d_in[0];
    const float* db = (const float*)d_in[1];
    const int B  = in_sizes[0] / 64;          // 256
    const int Nn = in_sizes[1] / 64;          // 1,000,000
    const int k  = 100;

    const int S = (Nn + NC - 1) / NC;         // #chunks

    auto align256 = [](size_t x) { return (x + 255) & ~(size_t)255; };
    const size_t candN    = (size_t)B * S * CAND;
    const size_t sz_xb    = align256((size_t)B * 64 * 2);
    const size_t sz_xsq   = align256((size_t)B * 4);
    const size_t sz_cd    = align256(candN * 4);
    const size_t sz_ci    = align256(candN * 4);
    const size_t sz_dbb   = align256((size_t)Nn * 64 * 2);
    const size_t sz_dbsq  = align256((size_t)Nn * 4);
    const size_t need_pre = sz_xb + sz_xsq + sz_cd + sz_ci + sz_dbb + sz_dbsq;

    const bool pre = (ws_size >= need_pre);

    char* ws = (char*)d_ws;
    size_t off = 0;
    auto carve = [&](size_t bytes) {
        void* p = ws + off;
        off += align256(bytes);
        return p;
    };
    unsigned short* Xb = (unsigned short*)carve((size_t)B * 64 * 2);
    float* xsq    = (float*)carve((size_t)B * 4);
    float* cand_d = (float*)carve(candN * 4);
    int*   cand_i = (int*)carve(candN * 4);
    unsigned short* dbb = nullptr;
    float* dbsq = nullptr;
    if (pre) {
        dbb  = (unsigned short*)carve((size_t)Nn * 64 * 2);
        dbsq = (float*)carve((size_t)Nn * 4);
    }

    float* out_d = (float*)d_out;
    int*   out_i = (int*)((float*)d_out + (size_t)B * k);

    x_prep<<<(B + TPB - 1) / TPB, TPB, 0, stream>>>(X, Xb, xsq, B);

    dim3 grid(S, B / QT);
    if (pre) {
        db_prep<<<((size_t)Nn * 4 + TPB - 1) / TPB, TPB, 0, stream>>>(db, dbb, dbsq, Nn);
        knn_main<true><<<grid, TPB, 0, stream>>>(Xb, xsq, db, dbb, dbsq,
                                                 cand_d, cand_i, Nn, S);
    } else {
        knn_main<false><<<grid, TPB, 0, stream>>>(Xb, xsq, db, nullptr, nullptr,
                                                  cand_d, cand_i, Nn, S);
    }

    const int C = S * CAND;
    topk_kernel<<<B, TPB, (size_t)C * 4, stream>>>(cand_d, cand_i, out_d, out_i, C, k);
}
